// QKVHadamard_40913858461802
// MI455X (gfx1250) — matrix operand order repaired and checked by reading, not compiled, not run on hardware
//
#include <hip/hip_runtime.h>

typedef __attribute__((ext_vector_type(16))) _Float16 v16h;
typedef __attribute__((ext_vector_type(8)))  _Float16 v8h;
typedef __attribute__((ext_vector_type(8)))  float    v8f;

#define B_ 2
#define N_ 512
#define D_ 512
#define E_ 64   // d_dir

union frag16 { v16h v; v8h h[2]; };

#if defined(__gfx1250__) && __has_builtin(__builtin_amdgcn_global_load_async_to_lds_b128) && __has_builtin(__builtin_amdgcn_s_wait_asynccnt)
#define HAVE_ASYNC_LDS 1
// Builtin signature (from hipcc diagnostic): arg0 = global ptr to int4-vector,
// arg1 = LDS ptr to int4-vector, arg2 = imm offset, arg3 = imm cpol.
typedef int v4i_t __attribute__((__vector_size__(16)));
typedef __attribute__((address_space(1))) v4i_t GV4;
typedef __attribute__((address_space(3))) v4i_t LV4;
__device__ __forceinline__ void async_copy_b128(const void* g, void* l) {
  // GLOBAL_LOAD_ASYNC_TO_LDS_B128: memory -> LDS direct, tracked by ASYNCcnt.
  __builtin_amdgcn_global_load_async_to_lds_b128((GV4*)g, (LV4*)l, 0, 0);
}
#endif

// ---------------------------------------------------------------------------
// Phase 1: q = t @ Wq^T (fp32), k = t @ Wk^T (stored fp16). Tiny (0.13 GFLOP).
// ---------------------------------------------------------------------------
__global__ __launch_bounds__(64) void qk_proj_kernel(
    const float* __restrict__ t, const float* __restrict__ Wq,
    const float* __restrict__ Wk, float* __restrict__ qout,
    _Float16* __restrict__ kout) {
  __shared__ float trow[D_];
  const int bn = blockIdx.x;        // b*N + n
  const int e  = threadIdx.x;       // 0..63
  const float* tp = t + (size_t)bn * D_;
  for (int d = e; d < D_; d += 64) trow[d] = tp[d];
  __syncthreads();
  const float* wq = Wq + (size_t)e * D_;
  const float* wk = Wk + (size_t)e * D_;
  float accq = 0.f, acck = 0.f;
  for (int d = 0; d < D_; ++d) {
    float tv = trow[d];
    accq = fmaf(tv, wq[d], accq);
    acck = fmaf(tv, wk[d], acck);
  }
  qout[(size_t)bn * E_ + e] = accq;
  kout[(size_t)bn * E_ + e] = (_Float16)acck;
}

// fp32 Wv -> fp16 ((D, d_dir) row-major == [d][e], B-operand-friendly layout).
__global__ __launch_bounds__(256) void wv_cvt_kernel(
    const float* __restrict__ Wv, _Float16* __restrict__ wvh) {
  int i = blockIdx.x * 256 + threadIdx.x;
  if (i < D_ * E_) wvh[i] = (_Float16)Wv[i];
}

// ---------------------------------------------------------------------------
// Phase 2: per (b,i): kappa[b,i,:,:] = (K_b * q_i) @ Wv^T   (512x64 @ 64x512)
// 256 threads = 8 wave32s. LDS: A (512x64 f16), W (512x64 f16), q_i (64 f32).
// Async-copy both 64KB operand panels into LDS, scale A by q_i in place,
// then per wave: hold 4 M-tile A fragments in registers, stream B fragments,
// two v_wmma_f32_16x16x32_f16 per 16x16 tile, NT stores of the 1 GiB output.
// ---------------------------------------------------------------------------
__global__ __launch_bounds__(256) void kappa_wmma_kernel(
    const float* __restrict__ qg, const _Float16* __restrict__ kh,
    const _Float16* __restrict__ wvh, float* __restrict__ out) {
  extern __shared__ char smem[];
  _Float16* ldsA = (_Float16*)smem;                 // 512*64 f16 = 64 KB
  _Float16* ldsW = (_Float16*)(smem + 64 * 1024);   // 512*64 f16 = 64 KB
  float*    qs   = (float*)   (smem + 128 * 1024);  // 64 f32

  const int bi  = blockIdx.x;       // b*N + i
  const int b   = bi >> 9;
  const int tid = threadIdx.x;

  if (tid < E_) qs[tid] = qg[(size_t)bi * E_ + tid];

  // Stage Wv panel and raw K_b panel (both f16, 4096 x 16B chunks each).
  const _Float16* kb = kh + (size_t)b * N_ * E_;
#ifdef HAVE_ASYNC_LDS
  for (int c = tid; c < (D_ * E_) / 8; c += 256)
    async_copy_b128((const char*)wvh + (size_t)c * 16, (char*)ldsW + (size_t)c * 16);
  for (int c = tid; c < (N_ * E_) / 8; c += 256)
    async_copy_b128((const char*)kb + (size_t)c * 16, (char*)ldsA + (size_t)c * 16);
  __builtin_amdgcn_s_wait_asynccnt(0);
#else
  {
    const uint4* ws = (const uint4*)wvh;
    const uint4* ks = (const uint4*)kb;
    uint4* wd = (uint4*)ldsW;
    uint4* kd = (uint4*)ldsA;
    for (int c = tid; c < (D_ * E_) / 8; c += 256) wd[c] = ws[c];
    for (int c = tid; c < (N_ * E_) / 8; c += 256) kd[c] = ks[c];
  }
#endif
  __syncthreads();

  // Scale A in place: A[j][e] *= q_i[e]  (f32 math, f16 result).
  for (int c = tid; c < (N_ * E_) / 8; c += 256) {
    const int e0 = (c & 7) * 8;
    v8h kv = *(v8h*)(ldsA + c * 8);
    v8h o;
#pragma unroll
    for (int u = 0; u < 8; ++u) o[u] = (_Float16)((float)kv[u] * qs[e0 + u]);
    *(v8h*)(ldsA + c * 8) = o;
  }
  __syncthreads();

  const int wave = tid >> 5;        // 0..7
  const int lane = tid & 31;
  const int g    = lane >> 4;       // ISA lane-group (K-half select)
  const int r    = lane & 15;

  float* outbase = out + (size_t)bi * N_ * D_;

  // Hold this wave's 4 M-tiles of A (rows wave*64 .. wave*64+63) in registers.
  frag16 a[4][2];
#pragma unroll
  for (int mt = 0; mt < 4; ++mt) {
    const int arow = ((wave * 4 + mt) * 16 + r) * E_;
    a[mt][0].h[0] = *(const v8h*)(ldsA + arow +      g * 8);
    a[mt][0].h[1] = *(const v8h*)(ldsA + arow + 16 + g * 8);
    a[mt][1].h[0] = *(const v8h*)(ldsA + arow + 32 + g * 8);
    a[mt][1].h[1] = *(const v8h*)(ldsA + arow + 48 + g * 8);
  }

  for (int nt = 0; nt < 32; ++nt) {
    const int n0   = nt * 16;
    const int wrow = (n0 + r) * E_ + g * 8;  // B: lane n, K = g*8 + s (s<8), 16 + g*8 + (s-8) (s>=8)
    frag16 b0, b1;
    b0.h[0] = *(const v8h*)(ldsW + wrow);
    b0.h[1] = *(const v8h*)(ldsW + wrow + 16);
    b1.h[0] = *(const v8h*)(ldsW + wrow + 32);
    b1.h[1] = *(const v8h*)(ldsW + wrow + 48);
#pragma unroll
    for (int mt = 0; mt < 4; ++mt) {
      const int m0 = (wave * 4 + mt) * 16;
      v8f c = {};
      c = __builtin_amdgcn_wmma_f32_16x16x32_f16(false, a[mt][0].v, false, b0.v,
                                                 (short)0, c, false, false);
      c = __builtin_amdgcn_wmma_f32_16x16x32_f16(false, a[mt][1].v, false, b1.v,
                                                 (short)0, c, false, false);
      // C/D layout: lane%16 = N column, VGPR rr -> M = m0 + rr + 8*g.
      float* op = outbase + (size_t)(m0 + 8 * g) * D_ + n0 + r;
#pragma unroll
      for (int rr = 0; rr < 8; ++rr)
        __builtin_nontemporal_store(c[rr], op + (size_t)rr * D_);
    }
  }
}

extern "C" void kernel_launch(void* const* d_in, const int* in_sizes, int n_in,
                              void* d_out, int out_size, void* d_ws, size_t ws_size,
                              hipStream_t stream) {
  const float* t  = (const float*)d_in[0];
  const float* Wq = (const float*)d_in[1];
  const float* Wk = (const float*)d_in[2];
  const float* Wv = (const float*)d_in[3];
  float* out = (float*)d_out;

  // Workspace: q fp32 (256 KB) | k fp16 (128 KB) | Wv fp16 (64 KB) = 448 KB.
  float*    qws = (float*)d_ws;
  _Float16* kws = (_Float16*)(qws + (size_t)B_ * N_ * E_);
  _Float16* wvh = kws + (size_t)B_ * N_ * E_;

  qk_proj_kernel<<<B_ * N_, 64, 0, stream>>>(t, Wq, Wk, qws, kws);
  wv_cvt_kernel<<<(D_ * E_ + 255) / 256, 256, 0, stream>>>(Wv, wvh);

  const size_t lds_bytes = 128 * 1024 + 256;  // A + W + q_i
  kappa_wmma_kernel<<<B_ * N_, 256, lds_bytes, stream>>>(qws, kws, wvh, out);
}
